// LSTM_19439021981778
// MI455X (gfx1250) — matrix operand order repaired and checked by reading, not compiled, not run on hardware
//
#include <hip/hip_runtime.h>
#include <hip/hip_bf16.h>
#include <stdint.h>

// ---------------------------------------------------------------------------
// 2-layer LSTM (S=512,B=64,In=256,H=1024) + FC, for MI455X (gfx1250, wave32).
// BF16 WMMA (16x16x32) with fp32 accumulation. Recurrent part runs as a
// persistent kernel: each of 128 workgroups owns 8 hidden units (32 gate
// columns), caches its [W_hh;W_ih] column slab in LDS for the whole sequence,
// bulk-stages the shared h vector into LDS each step (one high-MLP transfer
// instead of ~64 dependent L2 round trips), and a device-wide atomic barrier
// separates time steps.
// ---------------------------------------------------------------------------

#define SEQ    512
#define BATCH  64
#define IN_DIM 256
#define HID    1024
#define G4H    4096
#define OUTD   256
#define BSTRIDE 33    // weight-slab LDS row stride in words (pad vs conflicts)
#define HSTRIDE 1032  // staged-h LDS row stride in bf16 elems (516 words:
                      // lane i's b128 starts at bank 4i -> conflict-free)

typedef __attribute__((ext_vector_type(16))) __bf16    v16bf;
typedef __attribute__((ext_vector_type(8)))  float     v8f;
typedef __attribute__((ext_vector_type(4)))  uint32_t  v4u;

union FragBF { v4u q[2]; v16bf v; };

__device__ __forceinline__ uint16_t f32_to_bf16(float x) {
  union { float f; uint32_t u; } c; c.f = x;
  uint32_t r = c.u + 0x7FFFu + ((c.u >> 16) & 1u);   // round-to-nearest-even
  return (uint16_t)(r >> 16);
}

__device__ __forceinline__ float sigf(float x)   { return 1.0f / (1.0f + __expf(-x)); }
__device__ __forceinline__ float tanh_f(float x) { return 1.0f - 2.0f / (__expf(2.0f * x) + 1.0f); }

// A fragment: 16x32 bf16 tile, from a precomputed per-lane row pointer
// (works for both global and LDS row pointers; inlining specializes the AS).
// Lane L(0-15): M=L, K = base+{0..15}; lane L+16: M=L, K = base+{16..31}.
__device__ __forceinline__ v16bf load_a_frag(const uint16_t* __restrict__ rowp,
                                             int k_base, int koff) {
  const uint16_t* p = rowp + k_base + koff;
  FragBF f;
  f.q[0] = *reinterpret_cast<const v4u*>(p);        // K pairs 0..3   (b128)
  f.q[1] = *reinterpret_cast<const v4u*>(p + 8);    // K pairs 4..7   (b128)
  return f.v;
}

// B fragment: 32x16 bf16 tile from LDS packed K-pairs. bk is the loop-carried
// chunk pointer (lane column + lane-half row folded in); per-load offsets are
// v*33 words <= 231 -> fit ds_load_2addr 8-bit word-offset immediates.
__device__ __forceinline__ v16bf load_b_frag(const uint32_t* bk) {
  FragBF f;
#pragma unroll
  for (int v = 0; v < 8; ++v)
    f.q[v >> 2][v & 3] = bk[v * BSTRIDE];
  return f.v;
}

// ---------------------------------------------------------------------------
// Weight prep kernels
// ---------------------------------------------------------------------------

__global__ void cvt_bf16_kernel(const float* __restrict__ in,
                                uint16_t* __restrict__ out, int n) {
  int i = blockIdx.x * blockDim.x + threadIdx.x;
  if (i < n) out[i] = f32_to_bf16(in[i]);
}

// W [N,K] fp32 row-major -> packed bf16 pairs out[p][n] = {W[n,2p+1], W[n,2p]}
__global__ void pack_w_kernel(const float* __restrict__ W,
                              uint32_t* __restrict__ out, int N, int K) {
  int idx = blockIdx.x * blockDim.x + threadIdx.x;
  int total = N * (K >> 1);
  if (idx >= total) return;
  int p = idx / N, n = idx - p * N;
  uint32_t lo = f32_to_bf16(W[(size_t)n * K + 2 * p]);
  uint32_t hi = f32_to_bf16(W[(size_t)n * K + 2 * p + 1]);
  out[(size_t)p * N + n] = lo | (hi << 16);
}

__global__ void vecadd_kernel(const float* __restrict__ a,
                              const float* __restrict__ b,
                              float* __restrict__ o, int n) {
  int i = blockIdx.x * blockDim.x + threadIdx.x;
  if (i < n) o[i] = a[i] + b[i];
}

// ---------------------------------------------------------------------------
// Persistent recurrent LSTM layer.
// Grid: 128 WGs x 256 threads (8 waves). WG wg owns hidden units j0..j0+7,
// i.e. gate columns {g*1024 + j0 + jl : g<4, jl<8} -> 32 local columns.
// LDS (staged h FIRST so its per-chunk byte offsets fit DS immediates):
//   staged h [64 x 1032] bf16, B slab [(512 + K_in/2) rows x 32 cols] packed
//   bf16 pairs (rows 0..511 = W_hh K-pairs, rest = W_ih K-pairs),
//   gates [64 x 32] f32, c [64 x 8] f32.
// ---------------------------------------------------------------------------
__global__ void __launch_bounds__(256, 1)
lstm_layer_kernel(const uint16_t* __restrict__ x_seq,   // [S][B][K_in] bf16
                  const uint32_t* __restrict__ Whh_pk,  // [512][4096]
                  const uint32_t* __restrict__ Wih_pk,  // [K_in/2][4096]
                  const float*    __restrict__ bias,    // [4096]
                  uint16_t* __restrict__ h_buf0,        // [B][H] bf16 (zeroed)
                  uint16_t* __restrict__ h_buf1,        // [B][H] bf16
                  uint16_t* __restrict__ out_seq,       // [S][B][H] bf16 or null
                  int K_in,
                  int* __restrict__ sync_ctr)
{
  extern __shared__ char smem_raw[];
  const int tid = threadIdx.x;
  const int wg  = blockIdx.x;
  const int nwg = gridDim.x;
  const int j0  = wg * 8;
  const int Brows = 512 + (K_in >> 1);

  uint16_t* ldsH = reinterpret_cast<uint16_t*>(smem_raw);        // offset 0
  uint32_t* ldsB = reinterpret_cast<uint32_t*>(ldsH + 64 * HSTRIDE);
  float* gates   = reinterpret_cast<float*>(ldsB + (size_t)Brows * BSTRIDE);
  float* cst     = gates + 64 * BSTRIDE;   // [64][8] cell state

  // Stage this WG's 32 weight columns into LDS (resident all 512 steps).
  for (int idx = tid; idx < Brows * 32; idx += 256) {
    int p = idx >> 5, ln = idx & 31;
    int gcol = (ln >> 3) * HID + j0 + (ln & 7);
    uint32_t w = (p < 512) ? Whh_pk[(size_t)p * G4H + gcol]
                           : Wih_pk[(size_t)(p - 512) * G4H + gcol];
    ldsB[p * BSTRIDE + ln] = w;
  }
  for (int idx = tid; idx < 512; idx += 256) cst[idx] = 0.0f;

  const int lane  = tid & 31;
  const int wave  = tid >> 5;
  const int mtile = wave & 3;    // batch tile (4 x 16 = 64)
  const int ntile = wave >> 2;   // column tile (2 x 16 = 32)
  const int m     = lane & 15;
  const int koff  = (lane & 16) ? 16 : 0;
  const int hi8   = (lane & 16) ? 8 : 0;
  const int lncol = ntile * 16 + m;
  const float biasv = bias[(lncol >> 3) * HID + j0 + (lncol & 7)];
  const int Kx = K_in >> 5;      // 8 (L0) / 32 (L1) chunks for the x part

  // Hoisted per-wave invariant pointers.
  const uint16_t* ldsHrow = ldsH + (size_t)(mtile * 16 + m) * HSTRIDE;
  const uint32_t* bptr    = ldsB + (size_t)hi8 * BSTRIDE + lncol;

  __syncthreads();

  for (int t = 0; t < SEQ; ++t) {
    const uint16_t* h_rd = (t & 1) ? h_buf1 : h_buf0;
    uint16_t*       h_wr = (t & 1) ? h_buf0 : h_buf1;
    const uint16_t* xrow = x_seq + ((size_t)t * BATCH + mtile * 16 + m) * K_in;

    // Bulk-stage h_{t-1} [64][1024] bf16 into LDS: 8192 b128 segments,
    // 32 per thread, clause-batched -> one latency exposure per step.
    for (int i = tid; i < 8192; i += 256) {
      int row = i >> 7, seg = i & 127;
      *reinterpret_cast<v4u*>(ldsH + row * HSTRIDE + seg * 8) =
          *reinterpret_cast<const v4u*>(h_rd + row * HID + seg * 8);
    }
    __syncthreads();

    v8f acc;
#pragma unroll
    for (int r = 0; r < 8; ++r) acc[r] = biasv;   // fuse bias into C

    // gates[64,32] = h_{t-1} @ W_hh^T_cols + x_t @ W_ih^T_cols + b
    // bk: loop-carried B chunk pointer; flows from the h segment (ends at
    // 512 K-pairs) straight into the x segment.
    const uint32_t* bk = bptr;

    // h segment: all operands in LDS (short-latency ds loads).
#pragma unroll 4
    for (int kk = 0; kk < 32; ++kk) {
      v16bf a = load_a_frag(ldsHrow, kk * 32, koff);
      v16bf b = load_b_frag(bk);
      bk += 16 * BSTRIDE;
      acc = __builtin_amdgcn_wmma_f32_16x16x32_bf16(false, a, false, b,
                                                    (short)0, acc, false, false);
    }
    // x segment: A from global (prefetched into WGP cache last barrier).
#pragma unroll 4
    for (int kk = 0; kk < Kx; ++kk) {
      v16bf a = load_a_frag(xrow, kk * 32, koff);
      v16bf b = load_b_frag(bk);
      bk += 16 * BSTRIDE;
      acc = __builtin_amdgcn_wmma_f32_16x16x32_bf16(false, a, false, b,
                                                    (short)0, acc, false, false);
    }

#pragma unroll
    for (int r = 0; r < 8; ++r)
      gates[(mtile * 16 + hi8 + r) * BSTRIDE + lncol] = acc[r];

    __syncthreads();

    // Pointwise gate update for this WG's 8 hidden units x 64 batch.
    for (int e = tid; e < 512; e += 256) {
      int b  = e >> 3, jl = e & 7;
      float gi = gates[b * BSTRIDE + jl];
      float gf = gates[b * BSTRIDE + 8 + jl];
      float gg = gates[b * BSTRIDE + 16 + jl];
      float go = gates[b * BSTRIDE + 24 + jl];
      float c  = sigf(gf) * cst[e] + sigf(gi) * tanh_f(gg);
      cst[e] = c;
      float h  = sigf(go) * tanh_f(c);
      uint16_t hb = f32_to_bf16(h);
      h_wr[(size_t)b * HID + j0 + jl] = hb;
      if (out_seq) out_seq[((size_t)t * BATCH + b) * HID + j0 + jl] = hb;
    }

    __syncthreads();

    if (t + 1 < SEQ) {
      // Prefetch next step's x slab into cache while we sit in the barrier.
      {
        const char* nx = (const char*)(x_seq + (size_t)(t + 1) * BATCH * K_in);
        const int lines = (BATCH * K_in * 2) >> 7;   // 128B cachelines
        for (int i = tid; i < lines; i += 256)
          __builtin_prefetch(nx + ((size_t)i << 7), 0, 0);
      }
      // Device-wide step barrier: every WGP must see the full updated h.
      if (tid == 0) {
        __threadfence();
        atomicAdd(sync_ctr, 1);
        const int target = nwg * (t + 1);
        while (__hip_atomic_load(sync_ctr, __ATOMIC_RELAXED,
                                 __HIP_MEMORY_SCOPE_AGENT) < target)
          __builtin_amdgcn_s_sleep(1);
      }
      __syncthreads();
      __threadfence();
    }
  }
}

// ---------------------------------------------------------------------------
// FC: out[64,256] = h_last[64,1024](bf16) @ W_fc^T + b_fc   (fp32 out)
// Grid: 16 WGs x 128 threads (4 waves). WG owns 16 output columns; wave =
// batch tile. B fragments straight from L2-hot packed W_fc.
// ---------------------------------------------------------------------------
__global__ void __launch_bounds__(128, 1)
fc_kernel(const uint16_t* __restrict__ h,      // [64][1024] bf16
          const uint32_t* __restrict__ Wpk,    // [512][256] packed pairs
          const float*    __restrict__ bias,   // [256]
          float* __restrict__ out)             // [64][256]
{
  int tid   = threadIdx.x;
  int lane  = tid & 31;
  int mtile = tid >> 5;              // 0..3
  int n0    = blockIdx.x * 16;
  int m     = lane & 15;
  int koff  = (lane & 16) ? 16 : 0;
  int hi8   = (lane & 16) ? 8 : 0;

  const uint16_t* hrow = h + (size_t)(mtile * 16 + m) * HID;
  const uint32_t* wk   = Wpk + (size_t)hi8 * OUTD + n0 + m;  // carried pointer

  v8f acc;
#pragma unroll
  for (int r = 0; r < 8; ++r) acc[r] = 0.0f;

#pragma unroll 4
  for (int kk = 0; kk < 32; ++kk) {
    v16bf a = load_a_frag(hrow, kk * 32, koff);
    FragBF fb;
#pragma unroll
    for (int v = 0; v < 8; ++v)
      fb.q[v >> 2][v & 3] = wk[(size_t)v * OUTD];
    wk += 16 * OUTD;
    acc = __builtin_amdgcn_wmma_f32_16x16x32_bf16(false, a, false, fb.v,
                                                  (short)0, acc, false, false);
  }

  int col = n0 + m;
  float bv = bias[col];
#pragma unroll
  for (int r = 0; r < 8; ++r)
    out[(size_t)(mtile * 16 + hi8 + r) * OUTD + col] = acc[r] + bv;
}

// ---------------------------------------------------------------------------
// Host launch (graph-capture safe: only async stream ops + attribute set).
// Workspace layout (~112 MB assumed available in d_ws).
// ---------------------------------------------------------------------------
extern "C" void kernel_launch(void* const* d_in, const int* in_sizes, int n_in,
                              void* d_out, int out_size, void* d_ws, size_t ws_size,
                              hipStream_t stream) {
  (void)in_sizes; (void)n_in; (void)out_size; (void)ws_size;
  const float* x     = (const float*)d_in[0];
  const float* W_ih0 = (const float*)d_in[1];
  const float* W_hh0 = (const float*)d_in[2];
  const float* b_ih0 = (const float*)d_in[3];
  const float* b_hh0 = (const float*)d_in[4];
  const float* W_ih1 = (const float*)d_in[5];
  const float* W_hh1 = (const float*)d_in[6];
  const float* b_ih1 = (const float*)d_in[7];
  const float* b_hh1 = (const float*)d_in[8];
  const float* W_fc  = (const float*)d_in[9];
  const float* b_fc  = (const float*)d_in[10];

  char* ws = (char*)d_ws;
  size_t o = 0;
  auto carve = [&](size_t bytes) -> char* {
    char* p = ws + o; o += (bytes + 255) & ~(size_t)255; return p;
  };
  uint16_t* x0_bf   = (uint16_t*)carve((size_t)SEQ * BATCH * IN_DIM * 2);
  uint16_t* h1_seq  = (uint16_t*)carve((size_t)SEQ * BATCH * HID * 2);
  uint32_t* Whh0_pk = (uint32_t*)carve((size_t)512 * G4H * 4);
  uint32_t* Wih0_pk = (uint32_t*)carve((size_t)128 * G4H * 4);
  uint32_t* Whh1_pk = (uint32_t*)carve((size_t)512 * G4H * 4);
  uint32_t* Wih1_pk = (uint32_t*)carve((size_t)512 * G4H * 4);
  uint32_t* Wfc_pk  = (uint32_t*)carve((size_t)512 * OUTD * 4);
  float*    b0      = (float*)carve((size_t)G4H * 4);
  float*    b1      = (float*)carve((size_t)G4H * 4);
  uint16_t* h0a     = (uint16_t*)carve((size_t)BATCH * HID * 2);
  uint16_t* h0b     = (uint16_t*)carve((size_t)BATCH * HID * 2);
  uint16_t* h1a     = (uint16_t*)carve((size_t)BATCH * HID * 2);
  uint16_t* h1b     = (uint16_t*)carve((size_t)BATCH * HID * 2);
  int*      syncs   = (int*)carve(256);

  hipMemsetAsync(h0a,   0, (size_t)BATCH * HID * 2, stream);  // h_{-1} = 0
  hipMemsetAsync(h1a,   0, (size_t)BATCH * HID * 2, stream);
  hipMemsetAsync(syncs, 0, 256, stream);

  const int thr = 256;
  cvt_bf16_kernel<<<(SEQ * BATCH * IN_DIM + thr - 1) / thr, thr, 0, stream>>>(
      x, x0_bf, SEQ * BATCH * IN_DIM);
  pack_w_kernel<<<(512 * G4H + thr - 1) / thr, thr, 0, stream>>>(W_hh0, Whh0_pk, G4H, HID);
  pack_w_kernel<<<(128 * G4H + thr - 1) / thr, thr, 0, stream>>>(W_ih0, Wih0_pk, G4H, IN_DIM);
  pack_w_kernel<<<(512 * G4H + thr - 1) / thr, thr, 0, stream>>>(W_hh1, Whh1_pk, G4H, HID);
  pack_w_kernel<<<(512 * G4H + thr - 1) / thr, thr, 0, stream>>>(W_ih1, Wih1_pk, G4H, HID);
  pack_w_kernel<<<(512 * OUTD + thr - 1) / thr, thr, 0, stream>>>(W_fc, Wfc_pk, OUTD, HID);
  vecadd_kernel<<<(G4H + thr - 1) / thr, thr, 0, stream>>>(b_ih0, b_hh0, b0, G4H);
  vecadd_kernel<<<(G4H + thr - 1) / thr, thr, 0, stream>>>(b_ih1, b_hh1, b1, G4H);

  // LDS: staged h + weight slab + gates + cell state.
  const size_t smem0 = (size_t)64 * HSTRIDE * 2 +
                       (size_t)(512 + IN_DIM / 2) * BSTRIDE * 4 +
                       64 * BSTRIDE * 4 + 512 * 4;
  const size_t smem1 = (size_t)64 * HSTRIDE * 2 +
                       (size_t)(512 + HID / 2) * BSTRIDE * 4 +
                       64 * BSTRIDE * 4 + 512 * 4;
  hipFuncSetAttribute((const void*)lstm_layer_kernel,
                      hipFuncAttributeMaxDynamicSharedMemorySize, (int)smem1);

  // Layer 0: x (bf16) -> h1_seq; Layer 1: h1_seq -> final h (no seq output).
  lstm_layer_kernel<<<128, 256, smem0, stream>>>(
      x0_bf, Whh0_pk, Wih0_pk, b0, h0a, h0b, h1_seq, IN_DIM, syncs);
  lstm_layer_kernel<<<128, 256, smem1, stream>>>(
      h1_seq, Whh1_pk, Wih1_pk, b1, h1a, h1b, nullptr, HID, syncs + 64);

  // t=511 (odd) writes buffer 0 -> h_last lives in h1a.
  fc_kernel<<<16, 128, 0, stream>>>(h1a, Wfc_pk, b_fc, (float*)d_out);
}
